// Attention_49598282334847
// MI455X (gfx1250) — matrix operand order repaired and checked by reading, not compiled, not run on hardware
//
#include <hip/hip_runtime.h>
#include <hip/hip_fp16.h>

// ---------------- CDNA5 / gfx1250 WMMA types ----------------
typedef __attribute__((ext_vector_type(16))) _Float16 v16h;
typedef __attribute__((ext_vector_type(8)))  float    v8f;

#define DIMC   192
#define NHEADS 6
#define HD     32
#define NTOK   196
#define BWIN   512
#define NGRP   64
#define NQP    208          // 13*16 query rows padded
#define NKP    224          // 14*16 key rows padded (K-dim of P@V = 7*32)
#define MROWS  (BWIN*NTOK)  // 100352
#define MTILES (MROWS/16)   // 6272 (exact)

union V16 { v16h h; float4 q[2]; };

__device__ __forceinline__ v8f wmma16(v16h a, v16h b, v8f c) {
  // D = A(16x32 f16) * B(32x16 f16) + C(16x16 f32)
  return __builtin_amdgcn_wmma_f32_16x16x32_f16(false, a, false, b, (short)0, c,
                                                false, false);
}
// A-operand fragment: 16 contiguous halves at p (lane-half offset
// already folded into p by caller).
__device__ __forceinline__ void ld_a(V16& v, const _Float16* p) {
  v.q[0] = *(const float4*)p;
  v.q[1] = *(const float4*)(p + 8);
}
// B-operand fragment: 16 contiguous halves at p.
__device__ __forceinline__ void ld_b(V16& v, const _Float16* p) {
  v.q[0] = *(const float4*)p;
  v.q[1] = *(const float4*)(p + 8);
}

// ---------------- prep kernels ----------------
__global__ void k_f32_to_f16(const float* __restrict__ s, _Float16* __restrict__ d, long n) {
  long stride = (long)gridDim.x * blockDim.x;
  for (long i = (long)blockIdx.x * blockDim.x + threadIdx.x; i < n; i += stride)
    d[i] = (_Float16)s[i];
}

// biasM[h][m][n] = bias_table[rel_index[m*196+n]][h]
__global__ void k_bias(const float* __restrict__ table, const int* __restrict__ ridx,
                       float* __restrict__ biasM) {
  int i = blockIdx.x * blockDim.x + threadIdx.x;
  const int NN = NTOK * NTOK;
  if (i < NHEADS * NN) {
    int h  = i / NN;
    int mn = i - h * NN;
    biasM[i] = table[ridx[mn] * NHEADS + h];
  }
}

__global__ void k_zero_pads(_Float16* __restrict__ Qh, _Float16* __restrict__ Kh,
                            _Float16* __restrict__ Vh) {
  long stride = (long)gridDim.x * blockDim.x;
  long t0 = (long)blockIdx.x * blockDim.x + threadIdx.x;
  const long ZQ = (long)BWIN * NHEADS * (NQP - NTOK) * HD;
  const long ZK = (long)BWIN * NHEADS * (NKP - NTOK) * HD;
  for (long i = t0; i < ZQ; i += stride) {
    long bh = i / ((NQP - NTOK) * HD);
    long rem = i - bh * ((NQP - NTOK) * HD);
    long r = NTOK + rem / HD, c = rem % HD;
    Qh[(bh * NQP + r) * HD + c] = (_Float16)0.f;
  }
  for (long i = t0; i < ZK; i += stride) {
    long bh = i / ((NKP - NTOK) * HD);
    long rem = i - bh * ((NKP - NTOK) * HD);
    long r = NTOK + rem / HD, c = rem % HD;
    Kh[(bh * NKP + r) * HD + c] = (_Float16)0.f;
    Vh[(bh * NKP + r) * HD + c] = (_Float16)0.f;
  }
}

// ---------------- QKV projection GEMM ----------------
// 256 thr = 8 waves; each wave: one 16-row M-tile x 192 cols of (Q|K|V by grid.y)
__launch_bounds__(256)
__global__ void k_qkv(const _Float16* __restrict__ xh, const _Float16* __restrict__ wq,
                      _Float16* __restrict__ Qh, _Float16* __restrict__ Kh,
                      _Float16* __restrict__ Vh) {
  int lane = threadIdx.x & 31, wid = threadIdx.x >> 5;
  int l15 = lane & 15, hi = lane >> 4;
  int mtile = blockIdx.x * 8 + wid;
  int sel = blockIdx.y;
  const _Float16* wbase = wq + (size_t)sel * DIMC * DIMC;  // rows [sel*192, ...)
  v8f z = {0.f,0.f,0.f,0.f,0.f,0.f,0.f,0.f};
  v8f acc[12];
#pragma unroll
  for (int t = 0; t < 12; ++t) acc[t] = z;

  const _Float16* arow = xh + (size_t)(mtile * 16 + l15) * DIMC + hi * 16;
  const _Float16* brow = wbase + (size_t)l15 * DIMC + hi * 16;

  V16 A[2], B[2];
  ld_a(A[0], arow);
  ld_b(B[0], brow);
#pragma unroll
  for (int kc = 0; kc < 6; ++kc) {
    if (kc + 1 < 6) ld_a(A[(kc + 1) & 1], arow + (kc + 1) * 32);
#pragma unroll
    for (int t = 0; t < 12; ++t) {
      // prefetch next B tile while current WMMA consumes its buffer
      if (t + 1 < 12)
        ld_b(B[(t + 1) & 1], brow + (size_t)(t + 1) * 16 * DIMC + kc * 32);
      else if (kc + 1 < 6)
        ld_b(B[0], brow + (kc + 1) * 32);
      acc[t] = wmma16(A[kc & 1].h, B[t & 1].h, acc[t]);
    }
  }

  _Float16* dst = (sel == 0) ? Qh : ((sel == 1) ? Kh : Vh);
  int rp = (sel == 0) ? NQP : NKP;
  float scl = (sel == 0) ? 0.17677669529663689f : 1.0f;  // 1/sqrt(32) folded into Q
#pragma unroll
  for (int t = 0; t < 12; ++t) {
    int head = t >> 1;
    int dim = ((t & 1) << 4) + l15;
#pragma unroll
    for (int r = 0; r < 8; ++r) {
      int m = mtile * 16 + r + hi * 8;
      int b = m / NTOK, n = m - b * NTOK;
      dst[(((size_t)b * NHEADS + head) * rp + n) * HD + dim] = (_Float16)(acc[t][r] * scl);
    }
  }
}

// ---------------- fused attention per (b,h) ----------------
// 128 thr = 4 waves; wave handles query row-blocks rb, rb+4, ...
__launch_bounds__(128)
__global__ void k_attn(const _Float16* __restrict__ Qh, const _Float16* __restrict__ Kh,
                       const _Float16* __restrict__ Vh, const float* __restrict__ biasM,
                       const float* __restrict__ mask, _Float16* __restrict__ yh) {
  __shared__ _Float16 Kl[NKP * 40];       // K, row pitch 40 halves (bank-friendly)
  __shared__ _Float16 Vt[32 * 232];       // V transposed [dim][key], pitch 232
  __shared__ _Float16 Pl[4 * 16 * 232];   // per-wave P staging, pitch 232

  int lane = threadIdx.x & 31, wid = threadIdx.x >> 5;
  int l15 = lane & 15, hi = lane >> 4;
  int bh = blockIdx.x;
  int b = bh / NHEADS, h = bh - b * NHEADS;
  int g = b & (NGRP - 1);

  const _Float16* Kg = Kh + (size_t)bh * NKP * HD;
  const _Float16* Vg = Vh + (size_t)bh * NKP * HD;
  for (int i = threadIdx.x; i < NKP * HD; i += 128) {
    int r = i >> 5, c = i & 31;
    Kl[r * 40 + c] = Kg[i];
    Vt[c * 232 + r] = Vg[i];
  }
  __syncthreads();

  const _Float16* Qg = Qh + (size_t)bh * NQP * HD;
  const float* bm = biasM + (size_t)h * NTOK * NTOK;
  const float* mk = mask + (size_t)g * NTOK * NTOK;
  _Float16* Pw = &Pl[wid * 16 * 232];
  const _Float16* kbase = &Kl[(size_t)l15 * 40 + hi * 16];
  v8f z = {0.f,0.f,0.f,0.f,0.f,0.f,0.f,0.f};

  for (int rb = wid; rb < 13; rb += 4) {
    v8f S[14];
#pragma unroll
    for (int t = 0; t < 14; ++t) S[t] = z;

    V16 A;
    ld_a(A, Qg + (rb * 16 + l15) * HD + hi * 16);
    V16 Bb[2];
    ld_b(Bb[0], kbase);
#pragma unroll
    for (int t = 0; t < 14; ++t) {
      if (t + 1 < 14) ld_b(Bb[(t + 1) & 1], kbase + (t + 1) * 16 * 40);
      S[t] = wmma16(A.h, Bb[t & 1].h, S[t]);
    }

    int mrow = rb * 16 + hi * 8;
#pragma unroll
    for (int t = 0; t < 14; ++t) {
      int n = t * 16 + l15;
#pragma unroll
      for (int r = 0; r < 8; ++r) {
        int m = mrow + r;
        float v = -1e30f;
        if (n < NTOK && m < NTOK) v = S[t][r] + bm[m * NTOK + n] + mk[m * NTOK + n];
        S[t][r] = v;
      }
    }
    // row softmax: row lives in 16 lanes of one half-wave, one VGPR slot
    float inv[8], mx[8];
#pragma unroll
    for (int r = 0; r < 8; ++r) {
      float v = -1e30f;
#pragma unroll
      for (int t = 0; t < 14; ++t) v = fmaxf(v, S[t][r]);
      v = fmaxf(v, __shfl_xor(v, 1, 16));
      v = fmaxf(v, __shfl_xor(v, 2, 16));
      v = fmaxf(v, __shfl_xor(v, 4, 16));
      v = fmaxf(v, __shfl_xor(v, 8, 16));
      mx[r] = v;
    }
#pragma unroll
    for (int r = 0; r < 8; ++r) {
      float s = 0.f;
#pragma unroll
      for (int t = 0; t < 14; ++t) {
        float e = __expf(S[t][r] - mx[r]);
        S[t][r] = e;
        s += e;
      }
      s += __shfl_xor(s, 1, 16);
      s += __shfl_xor(s, 2, 16);
      s += __shfl_xor(s, 4, 16);
      s += __shfl_xor(s, 8, 16);
      inv[r] = (s > 0.f) ? 1.f / s : 0.f;
    }
    // stage P (C-layout -> LDS row-major), wave-private buffer, in-order LDS
#pragma unroll
    for (int t = 0; t < 14; ++t)
#pragma unroll
      for (int r = 0; r < 8; ++r)
        Pw[(r + hi * 8) * 232 + t * 16 + l15] = (_Float16)(S[t][r] * inv[r]);

    // O = P @ V  (K-dim 224 = 7 chunks of 32), pipelined LDS operand loads
    const _Float16* pbase = Pw + l15 * 232 + hi * 16;
    const _Float16* vbase0 = &Vt[(size_t)l15 * 232 + hi * 16];
    const _Float16* vbase1 = vbase0 + 16 * 232;
    v8f O0 = z, O1 = z;
    V16 A2[2], B0[2], B1[2];
    ld_a(A2[0], pbase);
    ld_b(B0[0], vbase0);
    ld_b(B1[0], vbase1);
#pragma unroll
    for (int kc = 0; kc < 7; ++kc) {
      int cur = kc & 1, nxt = cur ^ 1;
      if (kc + 1 < 7) {
        ld_a(A2[nxt], pbase + (kc + 1) * 32);
        ld_b(B0[nxt], vbase0 + (kc + 1) * 32);
        ld_b(B1[nxt], vbase1 + (kc + 1) * 32);
      }
      O0 = wmma16(A2[cur].h, B0[cur].h, O0);
      O1 = wmma16(A2[cur].h, B1[cur].h, O1);
    }
#pragma unroll
    for (int r = 0; r < 8; ++r) {
      int m = rb * 16 + r + hi * 8;
      if (m < NTOK) {
        size_t base = ((size_t)b * NTOK + m) * DIMC + h * HD;
        yh[base + l15]      = (_Float16)O0[r];
        yh[base + 16 + l15] = (_Float16)O1[r];
      }
    }
  }
}

// ---------------- output projection GEMM + bias ----------------
__launch_bounds__(256)
__global__ void k_proj(const _Float16* __restrict__ yh, const _Float16* __restrict__ wp,
                       const float* __restrict__ bp, float* __restrict__ out) {
  int lane = threadIdx.x & 31, wid = threadIdx.x >> 5;
  int l15 = lane & 15, hi = lane >> 4;
  int mtile = blockIdx.x * 8 + wid;
  v8f z = {0.f,0.f,0.f,0.f,0.f,0.f,0.f,0.f};
  v8f acc[12];
#pragma unroll
  for (int t = 0; t < 12; ++t) acc[t] = z;

  const _Float16* arow = yh + (size_t)(mtile * 16 + l15) * DIMC + hi * 16;
  const _Float16* brow = wp + (size_t)l15 * DIMC + hi * 16;

  V16 A[2], B[2];
  ld_a(A[0], arow);
  ld_b(B[0], brow);
#pragma unroll
  for (int kc = 0; kc < 6; ++kc) {
    if (kc + 1 < 6) ld_a(A[(kc + 1) & 1], arow + (kc + 1) * 32);
#pragma unroll
    for (int t = 0; t < 12; ++t) {
      if (t + 1 < 12)
        ld_b(B[(t + 1) & 1], brow + (size_t)(t + 1) * 16 * DIMC + kc * 32);
      else if (kc + 1 < 6)
        ld_b(B[0], brow + (kc + 1) * 32);
      acc[t] = wmma16(A[kc & 1].h, B[t & 1].h, acc[t]);
    }
  }
#pragma unroll
  for (int t = 0; t < 12; ++t) {
    float bb = bp[t * 16 + l15];
#pragma unroll
    for (int r = 0; r < 8; ++r) {
      int m = mtile * 16 + r + hi * 8;
      out[(size_t)m * DIMC + t * 16 + l15] = acc[t][r] + bb;
    }
  }
}

// ---------------- host launcher ----------------
extern "C" void kernel_launch(void* const* d_in, const int* in_sizes, int n_in,
                              void* d_out, int out_size, void* d_ws, size_t ws_size,
                              hipStream_t stream) {
  (void)in_sizes; (void)n_in; (void)out_size; (void)ws_size;
  const float* x     = (const float*)d_in[0];
  const float* wqkv  = (const float*)d_in[1];
  const float* wproj = (const float*)d_in[2];
  const float* bproj = (const float*)d_in[3];
  const float* btab  = (const float*)d_in[4];
  const float* mask  = (const float*)d_in[5];
  const int*   ridx  = (const int*)d_in[6];

  char* ws = (char*)d_ws;
  size_t off = 0;
  auto alloc = [&](size_t bytes) -> char* {
    char* p = ws + off;
    off = (off + bytes + 255) & ~(size_t)255;
    return p;
  };
  _Float16* xh  = (_Float16*)alloc((size_t)MROWS * DIMC * 2);   // also reused as yh
  _Float16* wqh = (_Float16*)alloc((size_t)3 * DIMC * DIMC * 2);
  _Float16* wph = (_Float16*)alloc((size_t)DIMC * DIMC * 2);
  float*    bM  = (float*)alloc((size_t)NHEADS * NTOK * NTOK * 4);
  _Float16* Qh  = (_Float16*)alloc((size_t)BWIN * NHEADS * NQP * HD * 2);
  _Float16* Kh  = (_Float16*)alloc((size_t)BWIN * NHEADS * NKP * HD * 2);
  _Float16* Vh  = (_Float16*)alloc((size_t)BWIN * NHEADS * NKP * HD * 2);
  _Float16* yh  = xh;  // xh is dead after k_qkv; reuse for attention output

  k_f32_to_f16<<<2048, 256, 0, stream>>>(x, xh, (long)MROWS * DIMC);
  k_f32_to_f16<<<128, 256, 0, stream>>>(wqkv, wqh, (long)3 * DIMC * DIMC);
  k_f32_to_f16<<<64, 256, 0, stream>>>(wproj, wph, (long)DIMC * DIMC);
  k_bias<<<(NHEADS * NTOK * NTOK + 255) / 256, 256, 0, stream>>>(btab, ridx, bM);
  k_zero_pads<<<256, 256, 0, stream>>>(Qh, Kh, Vh);

  dim3 gq(MTILES / 8, 3, 1);
  k_qkv<<<gq, 256, 0, stream>>>(xh, wqh, Qh, Kh, Vh);
  k_attn<<<BWIN * NHEADS, 128, 0, stream>>>(Qh, Kh, Vh, bM, mask, yh);
  k_proj<<<MTILES / 8, 256, 0, stream>>>(yh, wph, bproj, (float*)d_out);
}
